// RQ_36395552866388
// MI455X (gfx1250) — compile-verified, matrix operand order checked by reading, not run on hardware
//
#include <hip/hip_runtime.h>
#include <hip/hip_bf16.h>

typedef __attribute__((ext_vector_type(16))) __bf16 v16bf;
typedef __attribute__((ext_vector_type(8)))  __bf16 v8bf;
typedef __attribute__((ext_vector_type(8)))  float  v8f;

#define N_ROWS 65536
#define DDIM   256
#define KCODES 1024
#define LLEV   4
#define LOSS_SCALE (1.25f / 16777216.0f)   // (1+BETA)/(N*D), BETA=0.25

// ---------------- init: bf16 residual = data; zero loss slot ----------------
__global__ void rq_init(const float* __restrict__ data,
                        __bf16* __restrict__ r_bf,
                        float* __restrict__ loss_out) {
  size_t gid = (size_t)blockIdx.x * 256 + threadIdx.x;
  r_bf[gid] = (__bf16)data[gid];
  if (gid == 0) *loss_out = 0.0f;
}

// ------------- per-level codebook prep: bf16 convert + ||c||^2 --------------
__global__ void rq_prep_codebook(const float* __restrict__ cb,
                                 __bf16* __restrict__ cb_bf,
                                 float* __restrict__ cnorm) {
  const int k = blockIdx.x;
  const int d = threadIdx.x;
  float x = cb[(size_t)k * DDIM + d];
  cb_bf[(size_t)k * DDIM + d] = (__bf16)x;
  float s = x * x;
  #pragma unroll
  for (int off = 16; off >= 1; off >>= 1) s += __shfl_down(s, off, 32);
  __shared__ float sl[8];
  const int wid = d >> 5, lane = d & 31;
  if (lane == 0) sl[wid] = s;
  __syncthreads();
  if (d == 0) {
    float t = 0.0f;
    #pragma unroll
    for (int i = 0; i < 8; ++i) t += sl[i];
    cnorm[k] = t;
  }
}

// --------- fused bf16-WMMA GEMM (r @ cb^T) + rowwise argmin over K ----------
// Block: 256 threads = 8 waves, owns 64 rows. Waves tile 4(M) x 2(N).
// B tiles (32 codes x 256 D, 16 KB) are double-buffered in LDS via
// global_load_async_to_lds_b128 (ASYNCcnt) and shared by all 8 waves.
__global__ __launch_bounds__(256)
void rq_gemm_argmin(const __bf16* __restrict__ r_bf,
                    const __bf16* __restrict__ cb_bf,
                    const float* __restrict__ cnorm,
                    int* __restrict__ ids) {
  __shared__ __align__(16) __bf16 bt[2][32 * DDIM];   // 2 x 16 KB double buffer
  __shared__ float s_sc[2][64];
  __shared__ int   s_id[2][64];

  const int tid   = threadIdx.x;
  const int w     = tid >> 5;       // wave 0..7
  const int lane  = tid & 31;
  const int mw    = w >> 1;         // 0..3 (row quarter of the block)
  const int nw    = w & 1;          // 0..1 (code sub-tile)
  const int m0    = blockIdx.x * 64;
  const int rbase = m0 + mw * 16;
  const int lmod  = lane & 15;
  const int khalf = lane >> 4;

  // A fragments (16x32 bf16, ISA layout): lane holds row lmod, two contiguous
  // 8-elem runs per 32-wide K chunk. Resident across the whole code scan.
  v16bf av[8];
  {
    const __bf16* rowp = r_bf + (size_t)(rbase + lmod) * DDIM;
    #pragma unroll
    for (int dk = 0; dk < 8; ++dk) {
      v8bf lo = *(const v8bf*)(rowp + dk * 32 + khalf * 8);
      v8bf hi = *(const v8bf*)(rowp + dk * 32 + 16 + khalf * 8);
      v16bf a;
      #pragma unroll
      for (int i = 0; i < 8; ++i) { a[i] = lo[i]; a[i + 8] = hi[i]; }
      av[dk] = a;
    }
  }

  // Async-copy one 32-code B tile (16 KB) into LDS buffer `buf`.
  const uint64_t cb_base = (uint64_t)(uintptr_t)cb_bf;
  auto issue_tile = [&](int c, int buf) {
    const uint64_t g = cb_base + (uint64_t)c * (32 * DDIM * 2);
    #pragma unroll
    for (int i = 0; i < 4; ++i) {
      // Flat LDS addresses map via addr[31:0] -> truncation yields LDS offset.
      uint32_t lds = (uint32_t)(uintptr_t)(&bt[buf][0]) + i * 4096 + tid * 16;
      uint64_t ga  = g + (uint32_t)(i * 4096 + tid * 16);
      asm volatile("global_load_async_to_lds_b128 %0, %1, off"
                   :: "v"(lds), "v"(ga) : "memory");
    }
  };

  issue_tile(0, 0);
  asm volatile("s_wait_asynccnt 0" ::: "memory");
  __syncthreads();

  float best[8];
  int   bidx[8];
  #pragma unroll
  for (int v = 0; v < 8; ++v) { best[v] = 3.4e38f; bidx[v] = 0; }

  for (int c = 0; c < 32; ++c) {
    const int buf = c & 1;
    if (c + 1 < 32) issue_tile(c + 1, buf ^ 1);   // prefetch next tile

    // B fragments from LDS: lane holds column (code) nw*16 + lmod.
    const __bf16* colp = &bt[buf][(nw * 16 + lmod) * DDIM];
    v8f acc = {};
    #pragma unroll
    for (int dk = 0; dk < 8; ++dk) {
      v8bf lo = *(const v8bf*)(colp + dk * 32 + khalf * 8);
      v8bf hi = *(const v8bf*)(colp + dk * 32 + 16 + khalf * 8);
      v16bf b;
      #pragma unroll
      for (int i = 0; i < 8; ++i) { b[i] = lo[i]; b[i + 8] = hi[i]; }
      acc = __builtin_amdgcn_wmma_f32_16x16x32_bf16(
              false, av[dk], false, b, (short)0, acc, false, false);
    }

    // C/D layout: lane holds column n = lmod (within sub-tile); VGPR v -> row.
    const int   ncol = c * 32 + nw * 16 + lmod;
    const float cn   = cnorm[ncol];
    #pragma unroll
    for (int v = 0; v < 8; ++v) {
      float s = cn - 2.0f * acc[v];               // ||c||^2 - 2 r.c
      if (s < best[v]) { best[v] = s; bidx[v] = ncol; }
    }

    // Next tile landed + everyone done reading current buffer.
    asm volatile("s_wait_asynccnt 0" ::: "memory");
    __syncthreads();
  }

  // Min-reduce across the 16 lanes of each half-wave (the N dimension).
  #pragma unroll
  for (int step = 1; step < 16; step <<= 1) {
    #pragma unroll
    for (int v = 0; v < 8; ++v) {
      float os = __shfl_xor(best[v], step, 32);
      int   oi = __shfl_xor(bidx[v], step, 32);
      if (os < best[v] || (os == best[v] && oi < bidx[v])) {
        best[v] = os; bidx[v] = oi;
      }
    }
  }

  // Combine the 2 N-waves via LDS; rows are disjoint across the 4 M-waves.
  if (lmod == 0) {
    #pragma unroll
    for (int v = 0; v < 8; ++v) {
      int row = mw * 16 + khalf * 8 + v;
      s_sc[nw][row] = best[v];
      s_id[nw][row] = bidx[v];
    }
  }
  __syncthreads();
  if (tid < 64) {
    float bs = s_sc[0][tid];
    int   bi = s_id[0][tid];
    float os = s_sc[1][tid];
    int   oi = s_id[1][tid];
    if (os < bs || (os == bs && oi < bi)) { bs = os; bi = oi; }
    ids[m0 + tid] = bi;
  }
}

// ------- gather winning code, update zq_stack / residual / loss parts -------
__global__ void rq_update(const float* __restrict__ data,
                          const float* __restrict__ cb,
                          const int* __restrict__ ids,
                          const float* __restrict__ zq_prev,
                          float* __restrict__ zq_cur,
                          float* __restrict__ out_ids,
                          __bf16* __restrict__ r_bf,
                          float* __restrict__ partials,
                          int level) {
  const int n = blockIdx.x;           // one row per block
  const int d = threadIdx.x;
  const size_t gid = (size_t)n * DDIM + d;
  const int id = ids[n];
  const float q = cb[(size_t)id * DDIM + d];
  float z = q;
  if (level > 0) z += zq_prev[gid];
  zq_cur[gid] = z;
  const float rn = data[gid] - z;     // new residual = data - cumulative zq
  r_bf[gid] = (__bf16)rn;

  float val = rn * rn;
  #pragma unroll
  for (int off = 16; off >= 1; off >>= 1) val += __shfl_down(val, off, 32);
  __shared__ float sl[8];
  const int wid = d >> 5, lane = d & 31;
  if (lane == 0) sl[wid] = val;
  __syncthreads();
  if (d == 0) {
    float t = 0.0f;
    #pragma unroll
    for (int i = 0; i < 8; ++i) t += sl[i];
    partials[n] = t;
    out_ids[(size_t)n * LLEV + level] = (float)id;   // semantic id as float
  }
}

// -------- deterministic single-block loss accumulation (no atomics) ---------
__global__ void rq_loss_reduce(const float* __restrict__ partials,
                               float* __restrict__ loss_out) {
  const int t = threadIdx.x;
  float s = 0.0f;
  for (int i = t; i < N_ROWS; i += 256) s += partials[i];
  #pragma unroll
  for (int off = 16; off >= 1; off >>= 1) s += __shfl_down(s, off, 32);
  __shared__ float sl[8];
  if ((t & 31) == 0) sl[t >> 5] = s;
  __syncthreads();
  if (t == 0) {
    float tt = 0.0f;
    #pragma unroll
    for (int i = 0; i < 8; ++i) tt += sl[i];
    *loss_out += tt * LOSS_SCALE;
  }
}

extern "C" void kernel_launch(void* const* d_in, const int* in_sizes, int n_in,
                              void* d_out, int out_size, void* d_ws, size_t ws_size,
                              hipStream_t stream) {
  (void)in_sizes; (void)n_in; (void)out_size; (void)ws_size;

  const float* data      = (const float*)d_in[0];   // [65536, 256]
  const float* codebooks = (const float*)d_in[1];   // [4, 1024, 256]

  float* out     = (float*)d_out;
  float* zq      = out;                                         // [L,N,D]
  float* out_ids = out + (size_t)LLEV * N_ROWS * DDIM;          // [N,L]
  float* loss    = out_ids + (size_t)N_ROWS * LLEV;             // [1]

  char*   ws       = (char*)d_ws;
  __bf16* r_bf     = (__bf16*)ws;                               // N*D bf16 (32 MB)
  __bf16* cb_bf    = (__bf16*)(ws + (size_t)N_ROWS * DDIM * 2); // 512 KB
  float*  cnorm    = (float*)(ws + (size_t)N_ROWS * DDIM * 2
                                 + (size_t)KCODES * DDIM * 2);  // 4 KB
  int*    ids      = (int*)((char*)cnorm + KCODES * sizeof(float));      // 256 KB
  float*  partials = (float*)((char*)ids + (size_t)N_ROWS * sizeof(int)); // 256 KB

  rq_init<<<(N_ROWS * DDIM) / 256, 256, 0, stream>>>(data, r_bf, loss);

  for (int l = 0; l < LLEV; ++l) {
    const float* cb_l = codebooks + (size_t)l * KCODES * DDIM;
    rq_prep_codebook<<<KCODES, 256, 0, stream>>>(cb_l, cb_bf, cnorm);
    rq_gemm_argmin<<<N_ROWS / 64, 256, 0, stream>>>(r_bf, cb_bf, cnorm, ids);
    const float* zq_prev = (l == 0) ? zq : zq + (size_t)(l - 1) * N_ROWS * DDIM;
    rq_update<<<N_ROWS, 256, 0, stream>>>(data, cb_l, ids, zq_prev,
                                          zq + (size_t)l * N_ROWS * DDIM,
                                          out_ids, r_bf, partials, l);
    rq_loss_reduce<<<1, 256, 0, stream>>>(partials, loss);
  }
}
